// MultiHeadAttention_65240553226725
// MI455X (gfx1250) — matrix operand order repaired and checked by reading, not compiled, not run on hardware
//
#include <hip/hip_runtime.h>

#define BB  8
#define SS  1024
#define DD  1024
#define HH  16
#define DKK 64

typedef __attribute__((ext_vector_type(16))) _Float16 v16h;
typedef __attribute__((ext_vector_type(8)))  _Float16 v8h;
typedef __attribute__((ext_vector_type(8)))  float    v8f;

// ---- WMMA fragment helpers (contiguous k order: k = 16*(lane/16) + elem, wave32) ----

// A-matrix 16x32 f16: lane<16 holds row M=lane, elems 0..15 = K(k0..k0+15);
// lane>=16 holds row M=lane-16, K = k0+16..k0+31.
static __device__ inline v16h load_a_frag(const _Float16* base, int ld, int lane) {
  const int l = lane & 15, hi = lane >> 4;
  const _Float16* p = base + (size_t)l * ld + hi * 16;
  return *(const v16h*)p;
}

// B-matrix 32x16 f16 from N-major ("transposed") storage bt[n][k], stride ld:
// lane<16 = column n0+lane, K=k0..k0+15 ; lane>=16 = same column, K=k0+16..k0+31.
static __device__ inline v16h load_b_fragT(const _Float16* bt, int ld, int n0, int k0, int lane) {
  const int l = lane & 15, hi = lane >> 4;
  return *(const v16h*)(bt + (size_t)(n0 + l) * ld + k0 + hi * 16);
}

static __device__ inline v8f wmma16(v16h a, v16h b, v8f c) {
  return __builtin_amdgcn_wmma_f32_16x16x32_f16(false, a, false, b, (short)0, c, false, false);
}

// ---- CDNA5 async global->LDS copy (ASYNCcnt path) ----
typedef __attribute__((address_space(3))) const void* lds_cptr_t;

static __device__ inline uint32_t lds_addr(const void* p) {
  // generic -> LDS addrspace cast; ptrtoint yields the raw LDS byte offset.
  return (uint32_t)(uintptr_t)(lds_cptr_t)p;
}

static __device__ inline void async_copy16(uint32_t lds_off, const _Float16* gptr) {
  asm volatile("global_load_async_to_lds_b128 %0, %1, off"
               :: "v"(lds_off), "v"((uint64_t)(uintptr_t)gptr)
               : "memory");
}

static __device__ inline void wait_async0() {
  asm volatile("s_wait_asynccnt 0x0" ::: "memory");
}

// ---- conversion / transpose kernels ----

__global__ void k_cvt_x(const float* __restrict__ x, _Float16* __restrict__ xh, int n) {
  int i = blockIdx.x * blockDim.x + threadIdx.x;
  if (i < n) xh[i] = (_Float16)x[i];
}

// WT[w][h][n(dk)][k(d)] = W_w[h][d][dk]
__global__ void k_cvt_wqkv(const float* __restrict__ wq, const float* __restrict__ wk,
                           const float* __restrict__ wv, _Float16* __restrict__ wt) {
  int i = blockIdx.x * blockDim.x + threadIdx.x;
  const int per = HH * DD * DKK;
  if (i >= 3 * per) return;
  int w = i / per, r = i % per;
  int h = r / (DD * DKK); int r2 = r % (DD * DKK);
  int d = r2 / DKK, k = r2 % DKK;
  const float* src = (w == 0) ? wq : (w == 1) ? wk : wv;
  wt[(((size_t)w * HH + h) * DKK + k) * DD + d] =
      (_Float16)src[((size_t)h * DD + d) * DKK + k];
}

// WoT[n][k] = wo[k][n]
__global__ void k_cvt_wo(const float* __restrict__ wo, _Float16* __restrict__ wot) {
  int i = blockIdx.x * blockDim.x + threadIdx.x;
  if (i >= DD * DD) return;
  int k = i / DD, n = i % DD;
  wot[(size_t)n * DD + k] = (_Float16)wo[(size_t)k * DD + n];
}

__global__ void k_zero_stats(float* stats) {
  int i = threadIdx.x;
  if (i < 2 * BB) stats[i] = 0.f;
}

// ---- QKV projection: per (b, h, which) GEMM  [S x D] x [D x DK] ----
// 8 waves/block, M-tile 128, N = 64. B tile (64x32) staged in LDS via async
// copies, double-buffered; A fragments double-buffered in registers.
__global__ __launch_bounds__(256) void k_qkv(const _Float16* __restrict__ xh,
                                             const _Float16* __restrict__ wt,
                                             _Float16* __restrict__ qf,
                                             _Float16* __restrict__ kf,
                                             _Float16* __restrict__ vt) {
  __shared__ _Float16 bsm[2][64][32];  // 2 x 4KB B tiles
  const int tid = threadIdx.x;
  const int lane = tid & 31;
  const int wave = tid >> 5;
  const int h = blockIdx.y;
  const int which = blockIdx.z % 3;
  const int b = blockIdx.z / 3;
  const int m0 = blockIdx.x * 128 + wave * 16;

  const _Float16* abase = xh + ((size_t)b * SS + m0) * DD;
  const _Float16* bt = wt + (((size_t)which * HH + h) * DKK) * DD;

  // staging assignment: 256 threads cover 64 rows x 4 chunks of 16B
  const int sn = tid >> 2;
  const int sc = (tid & 3) * 8;  // halves

  // prologue: stage k0 = 0 into buffer 0; prefetch first A fragment
  async_copy16(lds_addr(&bsm[0][sn][sc]), bt + (size_t)sn * DD + sc);
  v16h a_cur = load_a_frag(abase, DD, lane);

  v8f c[4] = {};
  for (int k0 = 0; k0 < DD; k0 += 32) {
    const int buf = (k0 >> 5) & 1;
    wait_async0();       // my copy of tile k0 is in LDS
    __syncthreads();     // whole tile visible; all waves done with buf^1 reads
    v16h a_nxt = a_cur;
    if (k0 + 32 < DD) {
      async_copy16(lds_addr(&bsm[buf ^ 1][sn][sc]),
                   bt + (size_t)sn * DD + (k0 + 32) + sc);
      a_nxt = load_a_frag(abase + k0 + 32, DD, lane);
    }
    const _Float16* bl = &bsm[buf][0][0];
#pragma unroll
    for (int nt = 0; nt < 4; ++nt) {
      v16h bf = load_b_fragT(bl, 32, nt * 16, 0, lane);
      c[nt] = wmma16(a_cur, bf, c[nt]);
    }
    a_cur = a_nxt;
  }

  const int l = lane & 15, hi = lane >> 4;
  if (which == 2) {
    // store V transposed: Vt[b][h][dk][t]
    _Float16* dst = vt + ((size_t)(b * HH + h) * DKK) * SS;
#pragma unroll
    for (int nt = 0; nt < 4; ++nt)
#pragma unroll
      for (int r = 0; r < 8; ++r)
        dst[(size_t)(nt * 16 + l) * SS + m0 + r + hi * 8] = (_Float16)c[nt][r];
  } else {
    // Q/K row-major [b][h][s][dk]
    _Float16* dst = ((which == 0) ? qf : kf) + ((size_t)(b * HH + h) * SS + m0) * DKK;
#pragma unroll
    for (int nt = 0; nt < 4; ++nt)
#pragma unroll
      for (int r = 0; r < 8; ++r)
        dst[(size_t)(r + hi * 8) * DKK + nt * 16 + l] = (_Float16)c[nt][r];
  }
}

// ---- attention: one wave per (b, h, 16-query tile), pipelined ----
__global__ __launch_bounds__(32) void k_attn(const _Float16* __restrict__ qf,
                                             const _Float16* __restrict__ kf,
                                             const _Float16* __restrict__ vt,
                                             const int* __restrict__ mask,
                                             _Float16* __restrict__ ctxh) {
  __shared__ _Float16 at[16][SS];  // 32 KB: scores -> exp -> normalized attn
  const int lane = threadIdx.x;
  const int q0 = blockIdx.x * 16;
  const int h = blockIdx.y;
  const int b = blockIdx.z;
  const int l = lane & 15, hi = lane >> 4;

  const _Float16* qb = qf + ((size_t)(b * HH + h) * SS + q0) * DKK;
  const _Float16* kb = kf + ((size_t)(b * HH + h) * SS) * DKK;
  const _Float16* vb = vt + ((size_t)(b * HH + h) * DKK) * SS;
  const int* mrow = mask + ((size_t)b * SS + q0) * SS;

  // Q tile A-fragments held in registers for the whole kernel
  v16h aq0 = load_a_frag(qb, DKK, lane);
  v16h aq1 = load_a_frag(qb + 32, DKK, lane);

  // Phase 1: scores = Q K^T / 8, masked; K-tile B-frags + mask double-buffered
  v16h b0c = load_b_fragT(kb, DKK, 0, 0, lane);
  v16h b1c = load_b_fragT(kb, DKK, 0, 32, lane);
  int mv[8];
#pragma unroll
  for (int r = 0; r < 8; ++r) mv[r] = mrow[(size_t)(r + hi * 8) * SS + l];

  for (int t0 = 0; t0 < SS; t0 += 16) {
    v16h b0n = b0c, b1n = b1c;
    int mn[8];
#pragma unroll
    for (int r = 0; r < 8; ++r) mn[r] = mv[r];
    if (t0 + 16 < SS) {
      b0n = load_b_fragT(kb, DKK, t0 + 16, 0, lane);
      b1n = load_b_fragT(kb, DKK, t0 + 16, 32, lane);
#pragma unroll
      for (int r = 0; r < 8; ++r)
        mn[r] = mrow[(size_t)(r + hi * 8) * SS + t0 + 16 + l];
    }
    v8f c = {};
    c = wmma16(aq0, b0c, c);
    c = wmma16(aq1, b1c, c);
#pragma unroll
    for (int r = 0; r < 8; ++r) {
      int m = r + hi * 8;
      int t = t0 + l;
      float s = c[r] * 0.125f;  // 1/sqrt(64)
      at[m][t] = (_Float16)((mv[r] != 0) ? s : -65000.0f);
    }
    b0c = b0n; b1c = b1n;
#pragma unroll
    for (int r = 0; r < 8; ++r) mv[r] = mn[r];
  }
  __syncthreads();

  // Softmax: 2 lanes per row (lane l and l+16 split the 1024 columns)
  {
    const int row = l;
    const int c0 = hi * 512;
    float mx = -3.0e38f;
    for (int c = 0; c < 512; ++c) mx = fmaxf(mx, (float)at[row][c0 + c]);
    mx = fmaxf(mx, __shfl_xor(mx, 16, 32));
    float sum = 0.f;
    for (int c = 0; c < 512; ++c) {
      float e = __expf((float)at[row][c0 + c] - mx);
      at[row][c0 + c] = (_Float16)e;
      sum += e;
    }
    sum += __shfl_xor(sum, 16, 32);
    float inv = 1.f / sum;
    for (int c = 0; c < 512; ++c)
      at[row][c0 + c] = (_Float16)((float)at[row][c0 + c] * inv);
  }
  __syncthreads();

  // Phase 2: ctx = attn * V (reduction over t), A from LDS + B from global,
  // both double-buffered in registers.
  v8f cc[4] = {};
  v16h a_c = load_a_frag(&at[0][0], SS, lane);
  v16h bv_c[4];
#pragma unroll
  for (int nt = 0; nt < 4; ++nt) bv_c[nt] = load_b_fragT(vb, SS, nt * 16, 0, lane);

  for (int t0 = 0; t0 < SS; t0 += 32) {
    v16h a_n = a_c;
    v16h bv_n[4];
#pragma unroll
    for (int nt = 0; nt < 4; ++nt) bv_n[nt] = bv_c[nt];
    if (t0 + 32 < SS) {
      a_n = load_a_frag(&at[0][0] + t0 + 32, SS, lane);
#pragma unroll
      for (int nt = 0; nt < 4; ++nt)
        bv_n[nt] = load_b_fragT(vb, SS, nt * 16, t0 + 32, lane);
    }
#pragma unroll
    for (int nt = 0; nt < 4; ++nt) cc[nt] = wmma16(a_c, bv_c[nt], cc[nt]);
    a_c = a_n;
#pragma unroll
    for (int nt = 0; nt < 4; ++nt) bv_c[nt] = bv_n[nt];
  }
#pragma unroll
  for (int nt = 0; nt < 4; ++nt)
#pragma unroll
    for (int r = 0; r < 8; ++r)
      ctxh[((size_t)b * SS + q0 + r + hi * 8) * DD + h * DKK + nt * 16 + l] =
          (_Float16)cc[nt][r];
}

// ---- output projection + residual + batch sum/sumsq for layernorm ----
__global__ __launch_bounds__(256) void k_out(const _Float16* __restrict__ ctxh,
                                             const _Float16* __restrict__ wot,
                                             const float* __restrict__ x,
                                             float* __restrict__ out,
                                             float* __restrict__ stats) {
  __shared__ _Float16 bsm[2][64][32];  // 2 x 4KB B tiles (WoT)
  __shared__ float red[256];
  const int tid = threadIdx.x;
  const int lane = tid & 31;
  const int wave = tid >> 5;
  const int m0 = blockIdx.x * 128 + wave * 16;
  const int n0 = blockIdx.y * 64;
  const int l = lane & 15, hi = lane >> 4;

  const _Float16* abase = ctxh + (size_t)m0 * DD;
  const _Float16* bt = wot + (size_t)n0 * DD;

  const int sn = tid >> 2;
  const int sc = (tid & 3) * 8;

  async_copy16(lds_addr(&bsm[0][sn][sc]), bt + (size_t)sn * DD + sc);
  v16h a_cur = load_a_frag(abase, DD, lane);

  v8f c[4] = {};
  for (int k0 = 0; k0 < DD; k0 += 32) {
    const int buf = (k0 >> 5) & 1;
    wait_async0();
    __syncthreads();
    v16h a_nxt = a_cur;
    if (k0 + 32 < DD) {
      async_copy16(lds_addr(&bsm[buf ^ 1][sn][sc]),
                   bt + (size_t)sn * DD + (k0 + 32) + sc);
      a_nxt = load_a_frag(abase + k0 + 32, DD, lane);
    }
    const _Float16* bl = &bsm[buf][0][0];
#pragma unroll
    for (int nt = 0; nt < 4; ++nt) {
      v16h bf = load_b_fragT(bl, 32, nt * 16, 0, lane);
      c[nt] = wmma16(a_cur, bf, c[nt]);
    }
    a_cur = a_nxt;
  }

  float lsum = 0.f, lsq = 0.f;
#pragma unroll
  for (int nt = 0; nt < 4; ++nt)
#pragma unroll
    for (int r = 0; r < 8; ++r) {
      size_t idx = (size_t)(m0 + r + hi * 8) * DD + n0 + nt * 16 + l;
      float v = c[nt][r] + x[idx];
      out[idx] = v;
      lsum += v;
      lsq += v * v;
    }
  red[tid] = lsum; __syncthreads();
  for (int s = 128; s > 0; s >>= 1) { if (tid < s) red[tid] += red[tid + s]; __syncthreads(); }
  float bsum = red[0]; __syncthreads();
  red[tid] = lsq; __syncthreads();
  for (int s = 128; s > 0; s >>= 1) { if (tid < s) red[tid] += red[tid + s]; __syncthreads(); }
  if (tid == 0) {
    int bidx = blockIdx.x >> 3;  // 8 M-tiles per batch (1024 rows / 128)
    atomicAdd(&stats[2 * bidx], bsum);
    atomicAdd(&stats[2 * bidx + 1], red[0]);
  }
}

// ---- layernorm over (S, D) per batch, in place on d_out ----
__global__ void k_ln(float* __restrict__ out, const float* __restrict__ stats, int n) {
  int i = blockIdx.x * blockDim.x + threadIdx.x;
  if (i >= n) return;
  int bidx = i >> 20;  // / (S*D) = 1048576
  const float inv_n = 1.0f / (float)(SS * DD);
  float mean = stats[2 * bidx] * inv_n;
  float var = stats[2 * bidx + 1] * inv_n - mean * mean;
  out[i] = (out[i] - mean) * rsqrtf(var + 1e-5f);
}

extern "C" void kernel_launch(void* const* d_in, const int* in_sizes, int n_in,
                              void* d_out, int out_size, void* d_ws, size_t ws_size,
                              hipStream_t stream) {
  (void)in_sizes; (void)n_in; (void)out_size; (void)ws_size;
  const int*   mask = (const int*)d_in[0];
  const float* x    = (const float*)d_in[1];
  const float* wq   = (const float*)d_in[2];
  const float* wk   = (const float*)d_in[3];
  const float* wv   = (const float*)d_in[4];
  const float* wo   = (const float*)d_in[5];
  float* out = (float*)d_out;

  char* ws = (char*)d_ws;
  size_t off = 0;
  _Float16* Xh   = (_Float16*)(ws + off); off += (size_t)BB * SS * DD * 2;        // 16.8 MB
  _Float16* WT   = (_Float16*)(ws + off); off += (size_t)3 * HH * DKK * DD * 2;   //  6.3 MB
  _Float16* WoT  = (_Float16*)(ws + off); off += (size_t)DD * DD * 2;             //  2.1 MB
  _Float16* Qf   = (_Float16*)(ws + off); off += (size_t)BB * HH * SS * DKK * 2;  // 16.8 MB
  _Float16* Kf   = (_Float16*)(ws + off); off += (size_t)BB * HH * SS * DKK * 2;  // 16.8 MB
  _Float16* Vt   = (_Float16*)(ws + off); off += (size_t)BB * HH * SS * DKK * 2;  // 16.8 MB
  _Float16* Ctxh = (_Float16*)(ws + off); off += (size_t)BB * SS * DD * 2;        // 16.8 MB
  float*    stats = (float*)(ws + off);   off += 2 * BB * sizeof(float);

  const int n_x = BB * SS * DD;
  hipLaunchKernelGGL(k_cvt_x, dim3((n_x + 255) / 256), dim3(256), 0, stream, x, Xh, n_x);
  const int n_w = 3 * HH * DD * DKK;
  hipLaunchKernelGGL(k_cvt_wqkv, dim3((n_w + 255) / 256), dim3(256), 0, stream, wq, wk, wv, WT);
  hipLaunchKernelGGL(k_cvt_wo, dim3((DD * DD + 255) / 256), dim3(256), 0, stream, wo, WoT);
  hipLaunchKernelGGL(k_zero_stats, dim3(1), dim3(32), 0, stream, stats);
  hipLaunchKernelGGL(k_qkv, dim3(SS / 128, HH, BB * 3), dim3(256), 0, stream, Xh, WT, Qf, Kf, Vt);
  hipLaunchKernelGGL(k_attn, dim3(SS / 16, HH, BB), dim3(32), 0, stream, Qf, Kf, Vt, mask, Ctxh);
  hipLaunchKernelGGL(k_out, dim3(BB * SS / 128, DD / 64), dim3(256), 0, stream, Ctxh, WoT, x, out, stats);
  hipLaunchKernelGGL(k_ln, dim3((n_x + 255) / 256), dim3(256), 0, stream, out, stats, n_x);
}
